// Attention_model_63909113365106
// MI455X (gfx1250) — matrix operand order repaired and checked by reading, not compiled, not run on hardware
//
#include <hip/hip_runtime.h>

typedef __attribute__((ext_vector_type(16))) __bf16 v16bf;
typedef __attribute__((ext_vector_type(8)))  float  v8f;
typedef __attribute__((ext_vector_type(4)))  unsigned int u32x4;

union Frag16 { v16bf v; u32x4 q[2]; };

#define EMBED 256
#define UNITS 512
#define BATCH 2048
#define LOCS  64

// workspace layout (bytes); all offsets 256-aligned
#define OFF_H    0                      // fp32 [2048][512]  = 4 MB
#define OFF_W1T  4194304                // bf16 [512][256]   = 256 KB  (w1t[u][k] = W1[k][u])
#define OFF_W2T  4456448                // bf16 [512][512]   = 512 KB  (w2t[n][k] = W2[k][n])
#define OFF_HBF  4980736                // bf16 [2048][512]  = 2 MB
// total ~6.75 MB

// ---------------- kernel 0: fp32 -> bf16 conversions (+transposes) --------------
__global__ __launch_bounds__(256) void cvt_kernel(
    const float* __restrict__ W1, const float* __restrict__ W2,
    const float* __restrict__ hidden,
    __bf16* __restrict__ w1t, __bf16* __restrict__ w2t, __bf16* __restrict__ hbf) {
  int i = blockIdx.x * 256 + threadIdx.x;
  if (i < UNITS * EMBED) {                       // 131072: W1 transpose
    int u = i / EMBED, k = i % EMBED;
    w1t[i] = (__bf16)W1[k * UNITS + u];
  } else if (i < UNITS * EMBED + UNITS * UNITS) { // 262144: W2 transpose
    int j = i - UNITS * EMBED;
    int n = j / UNITS, k = j % UNITS;
    w2t[j] = (__bf16)W2[k * UNITS + n];
  } else if (i < UNITS * EMBED + UNITS * UNITS + BATCH * UNITS) {
    int h = i - (UNITS * EMBED + UNITS * UNITS);  // 1048576: hidden copy
    hbf[h] = (__bf16)hidden[h];
  }
}

// ---------------- kernel 1: H = hidden @ W2 + b2 (bf16 WMMA, fp32 out) ----------
__global__ __launch_bounds__(128) void projh_kernel(
    const __bf16* __restrict__ hbf, const __bf16* __restrict__ w2t,
    const float* __restrict__ b2, float* __restrict__ H) {
  const int lane  = threadIdx.x & 31;
  const int wid   = threadIdx.x >> 5;
  const int tile  = blockIdx.x * 4 + wid;   // 128 x 32 tiles
  const int m0    = (tile >> 5) << 4;
  const int n0    = (tile & 31) << 4;
  const int l16   = lane & 15;
  const int laneq = lane >> 4;

  // A: 16-bit A layout — lanes 0-15: K {0..7,16..23}; lanes 16-31: K {8..15,24..31}
  const __bf16* aptr = hbf + (m0 + l16) * UNITS + laneq * 8;
  // B: column-major per lane — lanes 0-15: K {0..7,16..23}; lanes 16-31: K {8..15,24..31}
  const __bf16* bptr = w2t + (n0 + l16) * UNITS + laneq * 8;

  v8f acc = {};
#pragma unroll
  for (int kk = 0; kk < 16; ++kk) {        // K = 512 / 32
    Frag16 a, bb;
    a.q[0]  = *reinterpret_cast<const u32x4*>(aptr + kk * 32);
    a.q[1]  = *reinterpret_cast<const u32x4*>(aptr + kk * 32 + 16);
    bb.q[0] = *reinterpret_cast<const u32x4*>(bptr + kk * 32);
    bb.q[1] = *reinterpret_cast<const u32x4*>(bptr + kk * 32 + 16);
    acc = __builtin_amdgcn_wmma_f32_16x16x32_bf16(false, a.v, false, bb.v,
                                                  (short)0, acc, false, false);
  }
  const int col = n0 + l16;
  const float badd = b2[col];
#pragma unroll
  for (int r = 0; r < 8; ++r)              // C layout: VGPR r -> M = r (+8 for upper lanes)
    H[(m0 + r + laneq * 8) * UNITS + col] = acc[r] + badd;
}

// ---------------- kernel 2: fused attention per batch element -------------------
__global__ __launch_bounds__(256) void attn_kernel(
    const float* __restrict__ features, const __bf16* __restrict__ w1t,
    const float* __restrict__ H, const float* __restrict__ b1,
    const float* __restrict__ V, float* __restrict__ out) {
  __shared__ __bf16 sFb[LOCS * 264];   // bf16 features tile, row pad 8 (528B stride)
  __shared__ float  hb[UNITS];         // H[b] + b1
  __shared__ float  vv[UNITS];         // V
  __shared__ float  score[LOCS];
  __shared__ float  wexp[LOCS];

  const int b = blockIdx.x;
  const int t = threadIdx.x;
  const float* fbase = features + b * (LOCS * EMBED);

  // stage features -> bf16 LDS (coalesced b128 loads, 64 elems/thread)
#pragma unroll
  for (int c = 0; c < 16; ++c) {
    int base = c * 1024 + t * 4;
    int row  = base >> 8;
    int k    = base & 255;
    float4 f = *reinterpret_cast<const float4*>(fbase + base);
    __bf16* d = &sFb[row * 264 + k];
    d[0] = (__bf16)f.x; d[1] = (__bf16)f.y; d[2] = (__bf16)f.z; d[3] = (__bf16)f.w;
  }
  {
    const float* Hrow = H + b * UNITS;
    hb[t]       = Hrow[t]       + b1[t];
    hb[t + 256] = Hrow[t + 256] + b1[t + 256];
    vv[t]       = V[t];
    vv[t + 256] = V[t + 256];
  }
  if (t < LOCS) score[t] = 0.f;
  __syncthreads();

  const int lane  = t & 31;
  const int wid   = t >> 5;
  const int m0    = (wid >> 1) << 4;   // 4 M-blocks of 16 rows
  const int nhalf = wid & 1;           // 2 N-halves of 256 cols
  const int l16   = lane & 15;
  const int laneq = lane >> 4;

  // load all A fragments (this wave's 16 rows x K=256) once; reuse across 16 N-tiles
  Frag16 afrag[8];
  {
    const __bf16* ap = &sFb[(m0 + l16) * 264 + laneq * 8];
#pragma unroll
    for (int kk = 0; kk < 8; ++kk) {
      afrag[kk].q[0] = *reinterpret_cast<const u32x4*>(ap + kk * 32);
      afrag[kk].q[1] = *reinterpret_cast<const u32x4*>(ap + kk * 32 + 16);
    }
  }

  float sc[8] = {0.f, 0.f, 0.f, 0.f, 0.f, 0.f, 0.f, 0.f};
  for (int nt = 0; nt < 16; ++nt) {
    const int n0 = nhalf * 256 + nt * 16;
    const int u  = n0 + l16;
    const __bf16* bp = w1t + u * EMBED + laneq * 8;
    v8f acc = {};
#pragma unroll
    for (int kk = 0; kk < 8; ++kk) {   // K = 256 / 32
      Frag16 bb;
      bb.q[0] = *reinterpret_cast<const u32x4*>(bp + kk * 32);
      bb.q[1] = *reinterpret_cast<const u32x4*>(bp + kk * 32 + 16);
      acc = __builtin_amdgcn_wmma_f32_16x16x32_bf16(false, afrag[kk].v, false, bb.v,
                                                    (short)0, acc, false, false);
    }
    // fused epilogue: e = tanh(acc + H[b][u] + b1[u]); sc += e * V[u]
    const float hbv = hb[u];
    const float vvl = vv[u];
#pragma unroll
    for (int r = 0; r < 8; ++r) {
      float ev = tanhf(acc[r] + hbv);
      sc[r] += ev * vvl;
    }
  }
  // reduce partial scores over the 16 lanes of each row group, then across waves via LDS
#pragma unroll
  for (int r = 0; r < 8; ++r) {
    float v = sc[r];
    v += __shfl_xor(v, 1);
    v += __shfl_xor(v, 2);
    v += __shfl_xor(v, 4);
    v += __shfl_xor(v, 8);
    if (l16 == 0) atomicAdd(&score[m0 + r + laneq * 8], v);
  }
  __syncthreads();

  // softmax over L=64 (bv cancels exactly); broadcast LDS reads are conflict-free
  float mx = -3.4e38f;
  for (int l = 0; l < LOCS; ++l) mx = fmaxf(mx, score[l]);
  if (t < LOCS) wexp[t] = __expf(score[t] - mx);
  __syncthreads();
  float denom = 0.f;
  for (int l = 0; l < LOCS; ++l) denom += wexp[l];
  const float inv = 1.0f / denom;

  // context[e] = sum_l w[l] * features[b][l][e]  (fp32, coalesced)
  float ctx = 0.f;
  const float* fe = fbase + t;
  for (int l = 0; l < LOCS; ++l) ctx += wexp[l] * fe[l * EMBED];
  out[b * EMBED + t] = ctx * inv;
  if (t < LOCS) out[BATCH * EMBED + b * LOCS + t] = wexp[t] * inv;
}

// ---------------------------------------------------------------------------------
extern "C" void kernel_launch(void* const* d_in, const int* in_sizes, int n_in,
                              void* d_out, int out_size, void* d_ws, size_t ws_size,
                              hipStream_t stream) {
  const float* features = (const float*)d_in[0];
  const float* hidden   = (const float*)d_in[1];
  const float* W1       = (const float*)d_in[2];
  const float* b1       = (const float*)d_in[3];
  const float* W2       = (const float*)d_in[4];
  const float* b2       = (const float*)d_in[5];
  const float* V        = (const float*)d_in[6];
  // d_in[7] = bv: additive constant on every logit -> cancels in softmax; unused.
  float* out = (float*)d_out;

  char* ws = (char*)d_ws;
  float*  H   = (float*)(ws + OFF_H);
  __bf16* w1t = (__bf16*)(ws + OFF_W1T);
  __bf16* w2t = (__bf16*)(ws + OFF_W2T);
  __bf16* hbf = (__bf16*)(ws + OFF_HBF);

  // 1,441,792 conversion elements / 256 = 5632 blocks
  cvt_kernel<<<5632, 256, 0, stream>>>(W1, W2, hidden, w1t, w2t, hbf);
  // 4096 tiles, 4 waves per 128-thread block
  projh_kernel<<<1024, 128, 0, stream>>>(hbf, w2t, b2, H);
  // one workgroup per batch element
  attn_kernel<<<2048, 256, 0, stream>>>(features, w1t, H, b1, V, out);
}
